// DeepSet_18356690223697
// MI455X (gfx1250) — matrix operand order repaired and checked by reading, not compiled, not run on hardware
//
#include <hip/hip_runtime.h>

typedef __attribute__((ext_vector_type(16))) __bf16        v16bf;
typedef __attribute__((ext_vector_type(8)))  float         v8f;
typedef __attribute__((ext_vector_type(8)))  unsigned int  v8u;

#define WB() asm volatile("" ::: "memory")

__device__ __forceinline__ unsigned int bfb(float x) {
  unsigned int u = __float_as_uint(x);
  u = (u + 0x7FFFu + ((u >> 16) & 1u)) >> 16;
  return u & 0xFFFFu;
}
__device__ __forceinline__ unsigned int pkbf(float a, float b) {
  return bfb(a) | (bfb(b) << 16);
}

__device__ __forceinline__ v8f wmma_bf16(v8u a, v8u b, v8f c) {
  union U { v8u u; v16bf h; } A, B;
  A.u = a; B.u = b;
  return __builtin_amdgcn_wmma_f32_16x16x32_bf16(
      false, A.h, false, B.h, (short)0, c, false, false);
}

// A-operand (16x32 bf16) from a row-major [*,256] bf16 row pointer (uint4 view).
// reg v: K = 32*c + 16*(v>=4) + 8*hi + 2*(v&3) (+1 in high half)
__device__ __forceinline__ v8u ldA128(const uint4* row, int hi, int c) {
  uint4 q0 = row[c * 4 + hi];
  uint4 q1 = row[c * 4 + hi + 2];
  v8u r;
  r[0] = q0.x; r[1] = q0.y; r[2] = q0.z; r[3] = q0.w;
  r[4] = q1.x; r[5] = q1.y; r[6] = q1.z; r[7] = q1.w;
  return r;
}

// B-operand tile from pre-packed stream: tile t occupies 256 dwords,
// lane L reads dwords [L*8, L*8+8).
__device__ __forceinline__ v8u ldB(const unsigned int* pk, int t, int lane) {
  const uint4* p = (const uint4*)(pk + (size_t)t * 256);
  uint4 q0 = p[lane * 2];
  uint4 q1 = p[lane * 2 + 1];
  v8u r;
  r[0] = q0.x; r[1] = q0.y; r[2] = q0.z; r[3] = q0.w;
  r[4] = q1.x; r[5] = q1.y; r[6] = q1.z; r[7] = q1.w;
  return r;
}

// RBF A-operand built on the fly: lane holds row M=lo (its own edge weight).
__device__ __forceinline__ v8u rbfA(float w, int c, int hi) {
  const float step  = 5.0f / 49.0f;
  const float coeff = -0.5f / (step * step);
  v8u r;
#pragma unroll
  for (int j = 0; j < 8; ++j) {
    int k = c * 32 + ((j >= 4) ? 16 : 0) + hi * 8 + (j & 3) * 2;
    float d0 = w - (float)k * step;
    float d1 = w - (float)(k + 1) * step;
    float a = (k     < 50) ? __expf(coeff * d0 * d0) : 0.f;
    float b = (k + 1 < 50) ? __expf(coeff * d1 * d1) : 0.f;
    r[j] = pkbf(a, b);
  }
  return r;
}

// ---------------- prep kernels ----------------

__global__ void zero_f32(float* p, size_t n) {
  size_t i = (size_t)blockIdx.x * blockDim.x + threadIdx.x;
  size_t s = (size_t)gridDim.x * blockDim.x;
  for (; i < n; i += s) p[i] = 0.f;
}

// x = emb[z] -> bf16 row-major [N][256] (stored as [N][128] dwords)
__global__ void gather_x(const int* __restrict__ z, const float* __restrict__ emb,
                         unsigned int* __restrict__ xout, int N) {
  size_t i = (size_t)blockIdx.x * blockDim.x + threadIdx.x;
  if (i >= (size_t)N * 128) return;
  int atom = (int)(i >> 7), p = (int)(i & 127);
  int zz = z[atom];
  const float* r = emb + (size_t)zz * 256 + p * 2;
  xout[i] = pkbf(r[0], r[1]);
}

// Pack W[Ksrc x Ncols] (row-major, K=in, N=out) into WMMA-B tile stream.
// B layout: reg j, lane L: K = kt*32 + 16*(j>=4) + 8*(L>=16) + 2*(j&3) (+1), N = nt*16 + (L&15)
__global__ void pack_plain(const float* __restrict__ W, unsigned int* __restrict__ out,
                           int Ksrc, int Ntiles, int Ncols) {
  int t = blockIdx.x, lane = threadIdx.x;
  int kt = t / Ntiles, nt = t - kt * Ntiles;
  int lo = lane & 15, hi = lane >> 4;
  int n = nt * 16 + lo;
  unsigned int* o = out + (size_t)t * 256 + lane * 8;
#pragma unroll
  for (int j = 0; j < 8; ++j) {
    int k = kt * 32 + ((j >= 4) ? 16 : 0) + hi * 8 + (j & 3) * 2;
    float a = (k     < Ksrc) ? W[(size_t)k * Ncols + n]       : 0.f;
    float b = (k + 1 < Ksrc) ? W[(size_t)(k + 1) * Ncols + n] : 0.f;
    o[j] = pkbf(a, b);
  }
}

// Combined gate weights: cols 0..9 = W_g, cols 16..25 = W_noise (each [256,10])
__global__ void pack_gate(const float* __restrict__ Wg, const float* __restrict__ Wn,
                          unsigned int* __restrict__ out) {
  int t = blockIdx.x, lane = threadIdx.x;
  int kt = t >> 1, nt = t & 1;
  int lo = lane & 15, hi = lane >> 4;
  const float* S = nt ? Wn : Wg;
  unsigned int* o = out + (size_t)t * 256 + lane * 8;
#pragma unroll
  for (int j = 0; j < 8; ++j) {
    int k = kt * 32 + ((j >= 4) ? 16 : 0) + hi * 8 + (j & 3) * 2;
    float a = (lo < 10) ? S[(size_t)k * 10 + lo]       : 0.f;
    float b = (lo < 10) ? S[(size_t)(k + 1) * 10 + lo] : 0.f;
    o[j] = pkbf(a, b);
  }
}

// W_exp [2,256,128] -> combined [256][256] (cols 0..127 expert0, 128..255 expert1)
__global__ void pack_exp(const float* __restrict__ We, unsigned int* __restrict__ out) {
  int t = blockIdx.x, lane = threadIdx.x;
  int kt = t >> 4, nt = t & 15;
  int lo = lane & 15, hi = lane >> 4;
  int n = nt * 16 + lo;
  int ex = n >> 7, f = n & 127;
  unsigned int* o = out + (size_t)t * 256 + lane * 8;
#pragma unroll
  for (int j = 0; j < 8; ++j) {
    int k = kt * 32 + ((j >= 4) ? 16 : 0) + hi * 8 + (j & 3) * 2;
    float a = We[(size_t)ex * 32768 + (size_t)k * 128 + f];
    float b = We[(size_t)ex * 32768 + (size_t)(k + 1) * 128 + f];
    o[j] = pkbf(a, b);
  }
}

// ---------------- main edge kernel ----------------

#define WAVES 2

__global__ __launch_bounds__(64, 1) void moe_edge_kernel(
    int E,
    const int* __restrict__ eidx, const float* __restrict__ ew,
    const float* __restrict__ evec, const float* __restrict__ noise,
    const float* __restrict__ bd, const float* __restrict__ bdt,
    const float* __restrict__ bai, const float* __restrict__ baj,
    const float* __restrict__ bgam, const float* __restrict__ bexp,
    const unsigned int* __restrict__ xbf,
    const unsigned int* __restrict__ pWd,  const unsigned int* __restrict__ pWdt,
    const unsigned int* __restrict__ pWai, const unsigned int* __restrict__ pWaj,
    const unsigned int* __restrict__ pWgam, const unsigned int* __restrict__ pWgate,
    const unsigned int* __restrict__ pWexp,
    float* __restrict__ outA, float* __restrict__ outV) {

  // Three bf16 staging buffers per wave: A (d_proj/a_i), B (d_t, later gamma), C (a_j)
  __shared__ __align__(16) unsigned short bufA[WAVES][16 * 256];
  __shared__ __align__(16) unsigned short bufB[WAVES][16 * 256];
  __shared__ __align__(16) unsigned short bufC[WAVES][16 * 256];
  __shared__ float Hs[WAVES][16 * 32];
  __shared__ float sC[WAVES][16];
  __shared__ float sW[WAVES][32];
  __shared__ float sEV[WAVES][48];
  __shared__ int   sSRC[WAVES][16];
  __shared__ int   sDST[WAVES][16];

  const int wave = threadIdx.x >> 5;
  const int lane = threadIdx.x & 31;
  const int lo = lane & 15, hi = lane >> 4;
  const int ntot = (E + 15) >> 4;
  const int tile = blockIdx.x * WAVES + wave;
  if (tile >= ntot) return;  // wave-uniform
  const int base = tile * 16;
  const int eg = (base + lo < E) ? (base + lo) : (E - 1);

  unsigned short* bA = bufA[wave];
  unsigned short* bB = bufB[wave];
  unsigned short* bC = bufC[wave];
  float* hs = Hs[wave];

  const uint4* rowA = (const uint4*)(bA + lo * 256);
  const uint4* rowB = (const uint4*)(bB + lo * 256);
  const uint4* rowC = (const uint4*)(bC + lo * 256);

  // per-edge scalars (lanes 0..15 own edge lo; hi half duplicates)
  float w = ew[eg];
  float Cc = (w < 5.0f) ? 0.5f * (__cosf(w * 0.62831853071795864769f) + 1.0f) : 0.0f;
  if (hi == 0) {
    sC[wave][lo] = Cc;
    float ex = evec[(size_t)eg * 3 + 0];
    float ey = evec[(size_t)eg * 3 + 1];
    float ez = evec[(size_t)eg * 3 + 2];
    float inv = 1.0f / sqrtf(ex * ex + ey * ey + ez * ez);
    sEV[wave][lo * 3 + 0] = ex * inv;
    sEV[wave][lo * 3 + 1] = ey * inv;
    sEV[wave][lo * 3 + 2] = ez * inv;
    sSRC[wave][lo] = eidx[eg];
    sDST[wave][lo] = eidx[E + eg];
  }
  WB();

  v8f acc[16];
  float bvs[16];

  // ---- S1: d_proj = RBF @ Wd, +bd, *C -> bufA ----
#pragma unroll
  for (int i = 0; i < 16; ++i) acc[i] = (v8f){0, 0, 0, 0, 0, 0, 0, 0};
#pragma unroll
  for (int c = 0; c < 2; ++c) {
    v8u a = rbfA(w, c, hi);
#pragma unroll
    for (int nt = 0; nt < 16; ++nt)
      acc[nt] = wmma_bf16(a, ldB(pWd, c * 16 + nt, lane), acc[nt]);
  }
#pragma unroll
  for (int nt = 0; nt < 16; ++nt) bvs[nt] = bd[nt * 16 + lo];
#pragma unroll
  for (int r = 0; r < 8; ++r) {
    int e = r + hi * 8;
    float cc = sC[wave][e];
#pragma unroll
    for (int nt = 0; nt < 16; ++nt)
      bA[e * 256 + nt * 16 + lo] = (unsigned short)bfb((acc[nt][r] + bvs[nt]) * cc);
  }
  WB();

  // ---- S2: d_t = d_proj @ Wdt + bdt -> bufB ----
#pragma unroll
  for (int i = 0; i < 16; ++i) acc[i] = (v8f){0, 0, 0, 0, 0, 0, 0, 0};
#pragma unroll
  for (int kt = 0; kt < 8; ++kt) {
    v8u a = ldA128(rowA, hi, kt);
#pragma unroll
    for (int nt = 0; nt < 16; ++nt)
      acc[nt] = wmma_bf16(a, ldB(pWdt, kt * 16 + nt, lane), acc[nt]);
  }
#pragma unroll
  for (int nt = 0; nt < 16; ++nt) bvs[nt] = bdt[nt * 16 + lo];
#pragma unroll
  for (int r = 0; r < 8; ++r) {
    int e = r + hi * 8;
#pragma unroll
    for (int nt = 0; nt < 16; ++nt)
      bB[e * 256 + nt * 16 + lo] = (unsigned short)bfb(acc[nt][r] + bvs[nt]);
  }
  WB();

  // ---- S2b: H = d_t @ [W_g | W_noise] ----
  {
    v8f h0 = (v8f){0, 0, 0, 0, 0, 0, 0, 0};
    v8f h1 = (v8f){0, 0, 0, 0, 0, 0, 0, 0};
#pragma unroll
    for (int kt = 0; kt < 8; ++kt) {
      v8u a = ldA128(rowB, hi, kt);
      h0 = wmma_bf16(a, ldB(pWgate, kt * 2 + 0, lane), h0);
      h1 = wmma_bf16(a, ldB(pWgate, kt * 2 + 1, lane), h1);
    }
#pragma unroll
    for (int r = 0; r < 8; ++r) {
      int e = r + hi * 8;
      hs[e * 32 + lo] = h0[r];
      hs[e * 32 + 16 + lo] = h1[r];
    }
  }
  WB();

  // ---- gating: noisy top-2 softmax (per-edge scalar work) ----
  {
    float Hv[10];
    size_t nb = (size_t)eg * 10;
#pragma unroll
    for (int g = 0; g < 10; ++g) {
      float a = hs[lo * 32 + g];
      float b = hs[lo * 32 + 16 + g];
      float sp = (b > 20.f) ? b : log1pf(__expf(b));
      Hv[g] = a + noise[nb + g] * sp;
    }
    float v0 = -3.4e38f, v1 = -3.4e38f;
#pragma unroll
    for (int g = 0; g < 10; ++g) {
      float h = Hv[g];
      if (h > v0) { v1 = v0; v0 = h; }
      else if (h > v1) { v1 = h; }
    }
    float S = 0.f;
#pragma unroll
    for (int g = 0; g < 10; ++g)
      if (Hv[g] >= v1) S += __expf(Hv[g] - v0);
    float w0g = 1.0f / S;
    float w1g = __expf(v1 - v0) / S;
    if (hi == 0) { sW[wave][lo * 2] = w0g; sW[wave][lo * 2 + 1] = w1g; }
  }
  WB();

  // ---- S3a: a_i = x[src] @ Wai + bai -> bufA (overwrites d_proj) ----
#pragma unroll
  for (int i = 0; i < 16; ++i) acc[i] = (v8f){0, 0, 0, 0, 0, 0, 0, 0};
  {
    const uint4* xr = (const uint4*)xbf + (size_t)sSRC[wave][lo] * 32;
#pragma unroll
    for (int kt = 0; kt < 8; ++kt) {
      v8u a = ldA128(xr, hi, kt);
#pragma unroll
      for (int nt = 0; nt < 16; ++nt)
        acc[nt] = wmma_bf16(a, ldB(pWai, kt * 16 + nt, lane), acc[nt]);
    }
  }
#pragma unroll
  for (int nt = 0; nt < 16; ++nt) bvs[nt] = bai[nt * 16 + lo];
#pragma unroll
  for (int r = 0; r < 8; ++r) {
    int e = r + hi * 8;
#pragma unroll
    for (int nt = 0; nt < 16; ++nt)
      bA[e * 256 + nt * 16 + lo] = (unsigned short)bfb(acc[nt][r] + bvs[nt]);
  }

  // ---- S3b: a_j = x[dst] @ Waj + baj -> bufC ----
#pragma unroll
  for (int i = 0; i < 16; ++i) acc[i] = (v8f){0, 0, 0, 0, 0, 0, 0, 0};
  {
    const uint4* xr = (const uint4*)xbf + (size_t)sDST[wave][lo] * 32;
#pragma unroll
    for (int kt = 0; kt < 8; ++kt) {
      v8u a = ldA128(xr, hi, kt);
#pragma unroll
      for (int nt = 0; nt < 16; ++nt)
        acc[nt] = wmma_bf16(a, ldB(pWaj, kt * 16 + nt, lane), acc[nt]);
    }
  }
#pragma unroll
  for (int nt = 0; nt < 16; ++nt) bvs[nt] = baj[nt * 16 + lo];
#pragma unroll
  for (int r = 0; r < 8; ++r) {
    int e = r + hi * 8;
#pragma unroll
    for (int nt = 0; nt < 16; ++nt)
      bC[e * 256 + nt * 16 + lo] = (unsigned short)bfb(acc[nt][r] + bvs[nt]);
  }
  WB();

  // ---- S4: gamma = [d_t | a_i | a_j] @ Wgam + bgam -> bufB (overwrites d_t) ----
#pragma unroll
  for (int i = 0; i < 16; ++i) acc[i] = (v8f){0, 0, 0, 0, 0, 0, 0, 0};
#pragma unroll
  for (int kt = 0; kt < 8; ++kt) {            // d_t rows 0..255 of Wgam
    v8u a = ldA128(rowB, hi, kt);
#pragma unroll
    for (int nt = 0; nt < 16; ++nt)
      acc[nt] = wmma_bf16(a, ldB(pWgam, kt * 16 + nt, lane), acc[nt]);
  }
#pragma unroll
  for (int kt = 0; kt < 8; ++kt) {            // a_i rows 256..511
    v8u a = ldA128(rowA, hi, kt);
#pragma unroll
    for (int nt = 0; nt < 16; ++nt)
      acc[nt] = wmma_bf16(a, ldB(pWgam, (8 + kt) * 16 + nt, lane), acc[nt]);
  }
#pragma unroll
  for (int kt = 0; kt < 8; ++kt) {            // a_j rows 512..767
    v8u a = ldA128(rowC, hi, kt);
#pragma unroll
    for (int nt = 0; nt < 16; ++nt)
      acc[nt] = wmma_bf16(a, ldB(pWgam, (16 + kt) * 16 + nt, lane), acc[nt]);
  }
  WB();
#pragma unroll
  for (int nt = 0; nt < 16; ++nt) bvs[nt] = bgam[nt * 16 + lo];
#pragma unroll
  for (int r = 0; r < 8; ++r) {
    int e = r + hi * 8;
#pragma unroll
    for (int nt = 0; nt < 16; ++nt)
      bB[e * 256 + nt * 16 + lo] = (unsigned short)bfb(acc[nt][r] + bvs[nt]);
  }
  WB();

  // ---- S5: experts: gamma @ [Wexp0 | Wexp1] ----
#pragma unroll
  for (int i = 0; i < 16; ++i) acc[i] = (v8f){0, 0, 0, 0, 0, 0, 0, 0};
#pragma unroll
  for (int kt = 0; kt < 8; ++kt) {
    v8u a = ldA128(rowB, hi, kt);
#pragma unroll
    for (int nt = 0; nt < 16; ++nt)
      acc[nt] = wmma_bf16(a, ldB(pWexp, kt * 16 + nt, lane), acc[nt]);
  }
  WB();

  // ---- S6: gate-combine + atomic scatter ----
#pragma unroll
  for (int r = 0; r < 8; ++r) {
    int e = r + hi * 8;
    if (base + e < E) {
      float w0g = sW[wave][e * 2], w1g = sW[wave][e * 2 + 1];
      int   at  = sSRC[wave][e];
      float vx = sEV[wave][e * 3 + 0];
      float vy = sEV[wave][e * 3 + 1];
      float vz = sEV[wave][e * 3 + 2];
      float* ab = outA + (size_t)at * 128;
      float* vb = outV + (size_t)at * 384;
#pragma unroll
      for (int nt = 0; nt < 8; ++nt) {
        int f = nt * 16 + lo;
        float eo = w0g * (acc[nt][r] + bexp[f]) +
                   w1g * (acc[nt + 8][r] + bexp[128 + f]);
        atomicAdd(ab + f, eo);
        atomicAdd(vb + f, vx * eo);
        atomicAdd(vb + 128 + f, vy * eo);
        atomicAdd(vb + 256 + f, vz * eo);
      }
    }
  }
}

// ---------------- launcher ----------------

extern "C" void kernel_launch(void* const* d_in, const int* in_sizes, int n_in,
                              void* d_out, int out_size, void* d_ws, size_t ws_size,
                              hipStream_t stream) {
  const int N = in_sizes[0];
  const int E = in_sizes[4];

  const int*   z    = (const int*)d_in[0];
  const int*   eidx = (const int*)d_in[3];
  const float* ew   = (const float*)d_in[4];
  const float* ev   = (const float*)d_in[5];
  const float* nz   = (const float*)d_in[6];
  const float* emb  = (const float*)d_in[7];
  const float* Wd   = (const float*)d_in[8];
  const float* bd   = (const float*)d_in[9];
  const float* Wdt  = (const float*)d_in[10];
  const float* bdt  = (const float*)d_in[11];
  const float* Wai  = (const float*)d_in[12];
  const float* bai  = (const float*)d_in[13];
  const float* Waj  = (const float*)d_in[14];
  const float* baj  = (const float*)d_in[15];
  const float* Wgam = (const float*)d_in[16];
  const float* bgam = (const float*)d_in[17];
  const float* Wg   = (const float*)d_in[18];
  const float* Wn   = (const float*)d_in[19];
  const float* Wexp = (const float*)d_in[20];
  const float* bexp = (const float*)d_in[21];

  char* ws = (char*)d_ws;
  size_t off = 0;
  auto take = [&](size_t bytes) {
    size_t o = off;
    off = (off + bytes + 255) & ~(size_t)255;
    return o;
  };
  unsigned int* xbf    = (unsigned int*)(ws + take((size_t)N * 256 * 2));
  unsigned int* pWd_   = (unsigned int*)(ws + take((size_t)32  * 1024));
  unsigned int* pWdt_  = (unsigned int*)(ws + take((size_t)128 * 1024));
  unsigned int* pWai_  = (unsigned int*)(ws + take((size_t)128 * 1024));
  unsigned int* pWaj_  = (unsigned int*)(ws + take((size_t)128 * 1024));
  unsigned int* pWgam_ = (unsigned int*)(ws + take((size_t)384 * 1024));
  unsigned int* pWgt_  = (unsigned int*)(ws + take((size_t)16  * 1024));
  unsigned int* pWex_  = (unsigned int*)(ws + take((size_t)128 * 1024));

  float* outA = (float*)d_out;
  float* outV = outA + (size_t)N * 128;

  zero_f32<<<4096, 256, 0, stream>>>((float*)d_out, (size_t)out_size);
  gather_x<<<(N * 128 + 255) / 256, 256, 0, stream>>>(z, emb, xbf, N);

  pack_plain<<<32,  32, 0, stream>>>(Wd,   pWd_,  50,  16, 256);
  pack_plain<<<128, 32, 0, stream>>>(Wdt,  pWdt_, 256, 16, 256);
  pack_plain<<<128, 32, 0, stream>>>(Wai,  pWai_, 256, 16, 256);
  pack_plain<<<128, 32, 0, stream>>>(Waj,  pWaj_, 256, 16, 256);
  pack_plain<<<384, 32, 0, stream>>>(Wgam, pWgam_, 768, 16, 256);
  pack_gate<<<16,  32, 0, stream>>>(Wg, Wn, pWgt_);
  pack_exp<<<128,  32, 0, stream>>>(Wexp, pWex_);

  int tiles = (E + 15) / 16;
  int blocks = (tiles + WAVES - 1) / WAVES;
  moe_edge_kernel<<<blocks, 64, 0, stream>>>(
      E, eidx, ew, ev, nz, bd, bdt, bai, baj, bgam, bexp, xbf,
      pWd_, pWdt_, pWai_, pWaj_, pWgam_, pWgt_, pWex_, outA, outV);
}
